// HGTLayer_90366111908555
// MI455X (gfx1250) — matrix operand order repaired and checked by reading, not compiled, not run on hardware
//
#include <hip/hip_runtime.h>
#include <math.h>

// ---------------------------------------------------------------------------
// Types for CDNA5 WMMA (gfx1250, wave32)
// ---------------------------------------------------------------------------
typedef __attribute__((ext_vector_type(16))) __bf16    v16bf;
typedef __attribute__((ext_vector_type(8)))  float     v8f;
typedef __attribute__((ext_vector_type(4)))  unsigned  v4u;

__device__ __forceinline__ unsigned pack_bf16x2(float x, float y) {
  // RNE fp32 -> bf16, packed pair (lo = x, hi = y)
  unsigned ux = __float_as_uint(x), uy = __float_as_uint(y);
  ux = (ux + 0x7FFFu + ((ux >> 16) & 1u)) >> 16;
  uy = (uy + 0x7FFFu + ((uy >> 16) & 1u)) >> 16;
  return (ux & 0xFFFFu) | (uy << 16);
}

__device__ __forceinline__ __bf16 tobf(float f) {
  unsigned u = __float_as_uint(f);
  unsigned r = (u + 0x7FFFu + ((u >> 16) & 1u)) >> 16;
  unsigned short s = (unsigned short)r;
  __bf16 b;
  __builtin_memcpy(&b, &s, 2);
  return b;
}

// ---------------------------------------------------------------------------
// Multi-B GEMM: O_m[M,128] = A[M,128] @ B_m[128,128]  for m < nmat (<=3)
// fp32 memory, bf16 WMMA compute, fp32 accumulate.
//
// Block = 256 threads = 8 waves; block stages its 128-row A stripe ONCE into
// LDS as packed bf16 (row stride 68 words = 272B: 16B-aligned fragment reads,
// bank-spread). Wave w owns column tile w. Per B-matrix the wave pre-loads all
// 4 K-step B fragments into registers, then sweeps 8 row tiles:
// 2x ds_load_b128 + 1 v_wmma per K-step.
//
// Fragment layouts per CDNA5 ISA 7.12.2 (16-bit operands):
//   A 16x32: lane half0 K=[0..7,16..23], half1 K=[8..15,24..31], row = lane%16
//   B 32x16: lane half0 K=[0..7,16..23], half1 K=[8..15,24..31], col = lane%16
//   C/D 16x16 f32: VGPR r -> row r (lanes 0-15) / r+8 (lanes 16-31), col=lane%16
// ---------------------------------------------------------------------------
#define LDS_STRIDE_W 68   // 32-bit words per staged row (64 data + 4 pad)

__global__ void wmma_gemm_128_multi(const float* __restrict__ A,
                                    const float* __restrict__ B0,
                                    const float* __restrict__ B1,
                                    const float* __restrict__ B2,
                                    float* __restrict__ O0,
                                    float* __restrict__ O1,
                                    float* __restrict__ O2,
                                    int M, int nmat) {
  __shared__ unsigned ldsA[128 * LDS_STRIDE_W];   // 34816 B of 320KB WGP LDS

  const int rbase = blockIdx.x * 128;

  // ---- Stage 128x128 fp32 stripe of A into LDS as packed bf16 ----
  for (int idx = threadIdx.x; idx < 128 * 64; idx += 256) {
    const int r  = idx >> 6;        // staged row 0..127
    const int cw = idx & 63;        // 32-bit word column (2 floats)
    unsigned packed = 0u;
    const int row = rbase + r;
    if (row < M) {
      const float2 f = ((const float2*)(A + (size_t)row * 128))[cw];
      packed = pack_bf16x2(f.x, f.y);
    }
    ldsA[r * LDS_STRIDE_W + cw] = packed;
  }
  __syncthreads();

  const int lane = threadIdx.x & 31;
  const int wave = threadIdx.x >> 5;   // 0..7 -> column tile
  const int col0 = wave * 16;
  const int r16  = lane & 15;          // fragment row / output col within tile
  const int kh   = lane >> 4;          // lane half selector

  const float* Bmats[3] = {B0, B1, B2};
  float*       Omats[3] = {O0, O1, O2};

  for (int m = 0; m < nmat; ++m) {
    const float* B = Bmats[m];
    float*       O = Omats[m];

    // ---- Pre-load B fragments for all 4 K-steps (held in registers) ----
    v16bf bfr[4];
    #pragma unroll
    for (int s = 0; s < 4; ++s) {
      const int k0 = s * 32;
      #pragma unroll
      for (int j = 0; j < 8; ++j) {
        const int Kb = (kh << 3) + 2 * j + (j >= 4 ? 8 : 0);
        const float* bp = B + (size_t)(k0 + Kb) * 128 + col0 + r16;
        bfr[s][2 * j]     = tobf(bp[0]);
        bfr[s][2 * j + 1] = tobf(bp[128]);
      }
    }

    // ---- Sweep 8 row tiles of the staged stripe ----
    #pragma unroll
    for (int t = 0; t < 8; ++t) {
      const int row0 = rbase + t * 16;
      if (row0 >= M) break;
      const unsigned* rowp = ldsA + (size_t)(t * 16 + r16) * LDS_STRIDE_W;

      v8f acc = {};
      #pragma unroll
      for (int s = 0; s < 4; ++s) {
        const int wbase = s * 16 + kh * 4;          // word offset: k0/2 + kh*4
        union { v16bf v; v4u u[2]; } ua;
        ua.u[0] = *(const v4u*)(rowp + wbase);      // K = k0+kh*8+{0..7}
        ua.u[1] = *(const v4u*)(rowp + wbase + 8);  // K = k0+kh*8+{16..23}
        acc = __builtin_amdgcn_wmma_f32_16x16x32_bf16(
            false, ua.v, false, bfr[s], (short)0, acc, false, false);
      }

      #pragma unroll
      for (int r = 0; r < 8; ++r) {
        const int row = row0 + r + (kh << 3);
        O[(size_t)row * 128 + col0 + r16] = acc[r];
      }
    }
  }
}

// ---------------------------------------------------------------------------
// Zero helper (grid-stride)
// ---------------------------------------------------------------------------
__global__ void zero_f32(float* __restrict__ p, size_t n) {
  size_t i = (size_t)blockIdx.x * blockDim.x + threadIdx.x;
  size_t stride = (size_t)gridDim.x * blockDim.x;
  for (; i < n; i += stride) p[i] = 0.0f;
}

// ---------------------------------------------------------------------------
// Phase 1: per-edge attention score + segment max (encoded-uint atomicMax).
// One wave32 per edge. k' = K[tail] @ relation_att[rel] per head;
// score = q.k'/sqrt(Dk). Broadcast of k[d] uses v_readlane (wave-uniform SGPR
// scalar source for v_fmac) -- no ds_bpermute round-trips.
// ---------------------------------------------------------------------------
__global__ void edge_score_kernel(const float* __restrict__ Qn, const float* __restrict__ Kn,
                                  const float* __restrict__ relAtt,
                                  const int* __restrict__ eHead, const int* __restrict__ eTail,
                                  const int* __restrict__ eType,
                                  float* __restrict__ score, unsigned* __restrict__ segmaxEnc,
                                  int E) {
  const int e = blockIdx.x * (blockDim.x >> 5) + (threadIdx.x >> 5);
  if (e >= E) return;
  const int lane = threadIdx.x & 31;
  const int hN = eHead[e], tN = eTail[e], rel = eType[e] - 1;
  const float rs = 0.17677669529663687f;  // 1/sqrt(32)

  #pragma unroll
  for (int h = 0; h < 4; ++h) {
    const float q  = Qn[(size_t)hN * 128 + h * 32 + lane];
    const float kv = Kn[(size_t)tN * 128 + h * 32 + lane];
    const float* Am = relAtt + (size_t)(rel * 4 + h) * 1024;  // [32][32]
    float kp = 0.0f;
    #pragma unroll
    for (int d = 0; d < 32; ++d) {
      const float kd = __builtin_amdgcn_readlane(kv, d);  // uniform scalar
      kp = fmaf(kd, Am[d * 32 + lane], kp);               // coalesced, L2-hot
    }
    float s = q * kp;
    #pragma unroll
    for (int off = 16; off; off >>= 1) s += __shfl_xor(s, off, 32);
    s *= rs;
    if (lane == 0) {
      score[(size_t)e * 4 + h] = s;
      const unsigned b = __float_as_uint(s);
      const unsigned enc = b ^ ((b & 0x80000000u) ? 0xFFFFFFFFu : 0x80000000u);
      atomicMax(&segmaxEnc[(size_t)hN * 4 + h], enc);
    }
  }
}

// ---------------------------------------------------------------------------
// Phase 2: ex = exp(score - segmax[head]); denom[head] += ex  (1 thread/(e,h))
// ---------------------------------------------------------------------------
__global__ void softmax_exp_kernel(const int* __restrict__ eHead,
                                   float* __restrict__ score,
                                   const unsigned* __restrict__ segmaxEnc,
                                   float* __restrict__ denom, int E4) {
  const int i = blockIdx.x * blockDim.x + threadIdx.x;
  if (i >= E4) return;
  const int e = i >> 2, h = i & 3;
  const int hN = eHead[e];
  const unsigned u = segmaxEnc[(size_t)hN * 4 + h];
  const unsigned bits = (u & 0x80000000u) ? (u & 0x7FFFFFFFu) : ~u;
  const float m = __uint_as_float(bits);
  const float ex = __expf(score[i] - m);
  score[i] = ex;
  atomicAdd(&denom[(size_t)hN * 4 + h], ex);
}

// ---------------------------------------------------------------------------
// Phase 3: v' = V[tail] @ relation_msg[rel]; aggPre[head] += attn * v'
// (W_O applied after segment-sum: segsum(X@W) == segsum(X)@W)
// ---------------------------------------------------------------------------
__global__ void edge_agg_kernel(const float* __restrict__ Vn, const float* __restrict__ relMsg,
                                const int* __restrict__ eHead, const int* __restrict__ eTail,
                                const int* __restrict__ eType,
                                const float* __restrict__ ex, const float* __restrict__ denom,
                                float* __restrict__ aggPre, int E) {
  const int e = blockIdx.x * (blockDim.x >> 5) + (threadIdx.x >> 5);
  if (e >= E) return;
  const int lane = threadIdx.x & 31;
  const int hN = eHead[e], tN = eTail[e], rel = eType[e] - 1;

  #pragma unroll
  for (int h = 0; h < 4; ++h) {
    const float vv = Vn[(size_t)tN * 128 + h * 32 + lane];
    const float* Mm = relMsg + (size_t)(rel * 4 + h) * 1024;
    float vp = 0.0f;
    #pragma unroll
    for (int d = 0; d < 32; ++d) {
      const float vd = __builtin_amdgcn_readlane(vv, d);  // uniform scalar
      vp = fmaf(vd, Mm[d * 32 + lane], vp);
    }
    const float a = ex[(size_t)e * 4 + h] / denom[(size_t)hN * 4 + h];
    atomicAdd(&aggPre[(size_t)hN * 128 + h * 32 + lane], vp * a);
  }
}

// ---------------------------------------------------------------------------
// User aggregation: out[row] += val * emb[col]  (one wave per nnz, float4 lanes)
// ---------------------------------------------------------------------------
__global__ void user_agg_kernel(const float* __restrict__ emb,
                                const int* __restrict__ rows, const int* __restrict__ cols,
                                const float* __restrict__ vals,
                                float* __restrict__ out, int NNZ) {
  const int i = blockIdx.x * (blockDim.x >> 5) + (threadIdx.x >> 5);
  if (i >= NNZ) return;
  const int lane = threadIdx.x & 31;
  const int r = rows[i], c = cols[i];
  const float v = vals[i];
  const float4 x = ((const float4*)(emb + (size_t)c * 128))[lane];
  float* dst = out + (size_t)r * 128 + lane * 4;
  atomicAdd(dst + 0, x.x * v);
  atomicAdd(dst + 1, x.y * v);
  atomicAdd(dst + 2, x.z * v);
  atomicAdd(dst + 3, x.w * v);
}

// ---------------------------------------------------------------------------
// Launch
// ---------------------------------------------------------------------------
extern "C" void kernel_launch(void* const* d_in, const int* in_sizes, int n_in,
                              void* d_out, int out_size, void* d_ws, size_t ws_size,
                              hipStream_t stream) {
  const float* emb    = (const float*)d_in[0];
  const float* W_Q    = (const float*)d_in[1];
  const float* W_K    = (const float*)d_in[2];
  const float* W_V    = (const float*)d_in[3];
  const float* W_O    = (const float*)d_in[4];
  const float* relAtt = (const float*)d_in[5];
  const float* relMsg = (const float*)d_in[6];
  const int*   eIdx   = (const int*)d_in[8];     // [2,E] row-major
  const int*   eType  = (const int*)d_in[9];
  const int*   imRows = (const int*)d_in[10];
  const int*   imCols = (const int*)d_in[11];
  const float* imVals = (const float*)d_in[12];

  const int N   = in_sizes[0] / 128;
  const int E   = in_sizes[9];
  const int NNZ = in_sizes[10];
  const int NU  = out_size / 128 - N;

  const int* eHead = eIdx;
  const int* eTail = eIdx + E;

  // Workspace layout (floats): Qn | Kn | Vn | aggPre | denom | segmaxEnc | score
  float* W = (float*)d_ws;
  const size_t NC = (size_t)N * 128;
  float*    Qn        = W;
  float*    Kn        = W + NC;
  float*    Vn        = W + 2 * NC;
  float*    aggPre    = W + 3 * NC;
  float*    denom     = W + 4 * NC;                 // N*4 floats
  unsigned* segmaxEnc = (unsigned*)(W + 4 * NC + (size_t)N * 4);
  float*    score     = W + 4 * NC + (size_t)N * 8; // E*4 floats

  float* outEnt  = (float*)d_out;                   // [N,128]
  float* outUser = (float*)d_out + NC;              // [NU,128]

  // 0) Zero atomic accumulation targets (every call: replays don't re-poison)
  zero_f32<<<2048, 256, 0, stream>>>(aggPre, NC + (size_t)N * 8);
  zero_f32<<<2048, 256, 0, stream>>>(outUser, (size_t)NU * 128);

  const int rowBlocks = (N + 127) / 128;

  // 1) Fused Q/K/V projections: A staged once in LDS, 3 WMMA GEMMs
  wmma_gemm_128_multi<<<rowBlocks, 256, 0, stream>>>(emb, W_Q, W_K, W_V,
                                                     Qn, Kn, Vn, N, 3);

  // 2) Edge scores + segment max
  edge_score_kernel<<<(E + 7) / 8, 256, 0, stream>>>(Qn, Kn, relAtt, eHead, eTail, eType,
                                                     score, segmaxEnc, E);

  // 3) exp + denom
  softmax_exp_kernel<<<(E * 4 + 255) / 256, 256, 0, stream>>>(eHead, score, segmaxEnc,
                                                              denom, E * 4);

  // 4) Relation-transformed V, attention-weighted segment sum (pre-W_O)
  edge_agg_kernel<<<(E + 7) / 8, 256, 0, stream>>>(Vn, relMsg, eHead, eTail, eType,
                                                   score, denom, aggPre, E);

  // 5) entity_agg = aggPre @ W_O  -> d_out[0 : N*128]
  wmma_gemm_128_multi<<<rowBlocks, 256, 0, stream>>>(aggPre, W_O, W_O, W_O,
                                                     outEnt, outEnt, outEnt, N, 1);

  // 6) user_agg SpMM -> d_out[N*128 : (N+NU)*128]
  user_agg_kernel<<<(NNZ + 7) / 8, 256, 0, stream>>>(emb, imRows, imCols, imVals,
                                                     outUser, NNZ);
}
